// RelPartialLearnableMultiHeadAttn_69587060130409
// MI455X (gfx1250) — matrix operand order repaired and checked by reading, not compiled, not run on hardware
//
#include <hip/hip_runtime.h>

typedef __bf16 bf16;
typedef bf16  v16bf __attribute__((ext_vector_type(16)));
typedef bf16  v8bf  __attribute__((ext_vector_type(8)));
typedef float v8f   __attribute__((ext_vector_type(8)));

#define QLEN   1024
#define MLEN   1024
#define KLEN   2048
#define BSZ    4
#define NH     16
#define DH     64
#define DMODEL 1024
#define SCALE  0.125f
#define LN_EPS 1e-5f

// ---------------- WMMA fragment helpers (wave32, 16x16x32 bf16) ----------------
// A-matrix 16x32 (MxK) per ISA 7.12.2: lane<16 -> K {0..7,16..23}; lane>=16 -> K {8..15,24..31}
static __device__ __forceinline__ v16bf ld_a16(const bf16* base, int ld) {
    int lane = threadIdx.x & 31;
    int m = lane & 15, h = (lane >> 4) & 1;
    const bf16* p = base + m * ld + h * 8;
    union { v16bf v; v8bf half[2]; } u;
    u.half[0] = *(const v8bf*)(p);
    u.half[1] = *(const v8bf*)(p + 16);
    return u.v;
}

// A fragment built from an fp32 buffer (LDS probabilities), converting to bf16 on the fly
static __device__ __forceinline__ v16bf ld_a16_f32(const float* base, int ld) {
    int lane = threadIdx.x & 31;
    int m = lane & 15, h = (lane >> 4) & 1;
    const float* p = base + m * ld + h * 8;
    v16bf f;
#pragma unroll
    for (int t = 0; t < 8; ++t) f[t] = (bf16)p[t];
#pragma unroll
    for (int t = 0; t < 8; ++t) f[t + 8] = (bf16)p[t + 16];
    return f;
}

// B-matrix 32x16 (KxN) from row-major Bn[N][K] (i.e. B = Bn^T):
// lane<16 -> K {0..7,16..23}; lane>=16 -> K {8..15,24..31}; N = lane&15
static __device__ __forceinline__ v16bf ld_bT16(const bf16* base, int ld) {
    int lane = threadIdx.x & 31;
    int n = lane & 15, h = (lane >> 4) & 1;
    const bf16* p = base + n * ld + h * 8;
    union { v16bf v; v8bf half[2]; } u;
    u.half[0] = *(const v8bf*)(p);
    u.half[1] = *(const v8bf*)(p + 16);
    return u.v;
}

static __device__ __forceinline__ v8f wmma_bf16(v16bf a, v16bf b, v8f c) {
    return __builtin_amdgcn_wmma_f32_16x16x32_bf16(false, a, false, b, (short)0, c,
                                                   false, false);
}

// ---------------- fp32 -> bf16 conversion ----------------
__global__ __launch_bounds__(256) void cvt_f32_bf16(const float* __restrict__ src,
                                                    bf16* __restrict__ dst, int n) {
    for (int i = blockIdx.x * blockDim.x + threadIdx.x; i < n;
         i += gridDim.x * blockDim.x)
        dst[i] = (bf16)src[i];
}

// ---------------- QKV projection: cat[8192,1024] @ Wqkv^T[1024,3072] -----------
// epilogue scatters into head-major K, transposed V, and two biased Q tensors
__global__ __launch_bounds__(256) void gemm_qkv(
    const bf16* __restrict__ cat, const bf16* __restrict__ Wqkv,
    const float* __restrict__ rwb, const float* __restrict__ rrb,
    bf16* __restrict__ qw, bf16* __restrict__ qr,
    bf16* __restrict__ kk, bf16* __restrict__ vvT) {
    const int NT = (3 * DMODEL) / 16;  // 192 col tiles
    int wave = threadIdx.x >> 5, lane = threadIdx.x & 31;
    int tile = blockIdx.x * 8 + wave;
    int mt = tile / NT, nt = tile % NT;
    const bf16* A0 = cat + mt * 16 * DMODEL;
    const bf16* B0 = Wqkv + nt * 16 * DMODEL;
    v8f acc = {};
    for (int k0 = 0; k0 < DMODEL; k0 += 32) {
        v16bf a = ld_a16(A0 + k0, DMODEL);
        v16bf b = ld_bT16(B0 + k0, DMODEL);
        acc = wmma_bf16(a, b, acc);
    }
    int n16 = lane & 15, h = lane >> 4;
    int col = nt * 16 + n16;
    int part = col >> 10, rem = col & 1023;
    int hd = rem >> 6, d = rem & 63;
#pragma unroll
    for (int p = 0; p < 8; ++p) {
        int row = mt * 16 + p + 8 * h;   // (t, b) flattened: row = t*BSZ + b
        int t = row >> 2, b_ = row & 3;
        float v = acc[p];
        if (part == 1) {
            kk[((b_ * NH + hd) * KLEN + t) * DH + d] = (bf16)v;
        } else if (part == 2) {
            vvT[((b_ * NH + hd) * DH + d) * KLEN + t] = (bf16)v;
        } else {
            int t2 = t - MLEN;
            if (t2 >= 0) {
                qw[((b_ * NH + hd) * QLEN + t2) * DH + d] = (bf16)(v + rwb[hd * DH + d]);
                qr[((b_ * NH + hd) * QLEN + t2) * DH + d] = (bf16)(v + rrb[hd * DH + d]);
            }
        }
    }
}

// ---------------- R projection: r[2048,1024] @ Wr^T -> head-major rk ----------
__global__ __launch_bounds__(256) void gemm_r(const bf16* __restrict__ rbf,
                                              const bf16* __restrict__ Wr,
                                              bf16* __restrict__ rk) {
    const int NT = DMODEL / 16;  // 64
    int wave = threadIdx.x >> 5, lane = threadIdx.x & 31;
    int tile = blockIdx.x * 8 + wave;
    int mt = tile / NT, nt = tile % NT;
    v8f acc = {};
    for (int k0 = 0; k0 < DMODEL; k0 += 32) {
        v16bf a = ld_a16(rbf + mt * 16 * DMODEL + k0, DMODEL);
        v16bf b = ld_bT16(Wr + nt * 16 * DMODEL + k0, DMODEL);
        acc = wmma_bf16(a, b, acc);
    }
    int n16 = lane & 15, h = lane >> 4;
    int col = nt * 16 + n16;
    int hd = col >> 6, d = col & 63;
#pragma unroll
    for (int p = 0; p < 8; ++p) {
        int j = mt * 16 + p + 8 * h;
        rk[(hd * KLEN + j) * DH + d] = (bf16)acc[p];
    }
}

// ---------------- fused attention core --------------------------------------
// one block = (b, head, 16-row q tile); scores live entirely in LDS
__global__ __launch_bounds__(256) void attn_kernel(
    const bf16* __restrict__ qw, const bf16* __restrict__ qr,
    const bf16* __restrict__ kk, const bf16* __restrict__ rk,
    const bf16* __restrict__ vvT, bf16* __restrict__ av) {
    __shared__ __align__(32) float S[16 * KLEN];   // 128 KB score/prob buffer
    __shared__ __align__(32) bf16 Qw[16 * DH];
    __shared__ __align__(32) bf16 Qr[16 * DH];
    __shared__ __align__(32) float O[16 * DH];
    int tid = threadIdx.x, wave = tid >> 5, lane = tid & 31;
    int bid = blockIdx.x;
    int it = bid & 63, hn = (bid >> 6) & 15, b = bid >> 10;
    int i0 = it * 16;

    const bf16* qwp = qw + ((b * NH + hn) * QLEN + i0) * DH;
    const bf16* qrp = qr + ((b * NH + hn) * QLEN + i0) * DH;
    for (int e = tid; e < 16 * DH; e += 256) {
        Qw[e] = qwp[e];
        Qr[e] = qrp[e];
        O[e] = 0.f;
    }
    __syncthreads();

    // Phase 1: S = AC = Qw(16x64) @ K^T(64x2048)
    const bf16* kbase = kk + (b * NH + hn) * KLEN * DH;
    for (int jt = wave; jt < KLEN / 16; jt += 8) {
        int j0 = jt * 16;
        v8f acc = {};
#pragma unroll
        for (int kc = 0; kc < 2; ++kc) {
            v16bf a = ld_a16(Qw + kc * 32, DH);
            v16bf bb = ld_bT16(kbase + j0 * DH + kc * 32, DH);
            acc = wmma_bf16(a, bb, acc);
        }
        int n16 = lane & 15, h = lane >> 4;
#pragma unroll
        for (int p = 0; p < 8; ++p) S[(p + 8 * h) * KLEN + j0 + n16] = acc[p];
    }
    __syncthreads();

    // Phase 2: BDraw tiles, scatter-added with the rel_shift fold:
    // BD[i,j] = BDraw[i, j + QLEN-1-i]; shift's zero region == causal mask region
    const bf16* rbase = rk + hn * KLEN * DH;
    for (int jt = wave; jt < KLEN / 16; jt += 8) {
        int j0 = jt * 16;
        v8f acc = {};
#pragma unroll
        for (int kc = 0; kc < 2; ++kc) {
            v16bf a = ld_a16(Qr + kc * 32, DH);
            v16bf bb = ld_bT16(rbase + j0 * DH + kc * 32, DH);
            acc = wmma_bf16(a, bb, acc);
        }
        int n16 = lane & 15, h = lane >> 4;
#pragma unroll
        for (int p = 0; p < 8; ++p) {
            int li = p + 8 * h;
            int i = i0 + li;
            int j = (j0 + n16) - (QLEN - 1 - i);
            if (j >= 0) S[li * KLEN + j] += acc[p];  // unique (li,j) -> race-free
        }
    }
    __syncthreads();

    // Phase 3: masked softmax per row (one wave owns rows wave*2, wave*2+1)
    for (int rr = 0; rr < 2; ++rr) {
        int li = wave * 2 + rr;
        int i = i0 + li;
        int jmax = i + MLEN;  // mask: j > i + MLEN
        float* Sr = S + li * KLEN;
        float m = -1e30f;
        for (int j = lane; j <= jmax; j += 32) m = fmaxf(m, Sr[j]);
#pragma unroll
        for (int off = 16; off; off >>= 1) m = fmaxf(m, __shfl_xor(m, off));
        float sm = m * SCALE;
        float sum = 0.f;
        for (int j = lane; j < KLEN; j += 32) {
            float e = 0.f;
            if (j <= jmax) { e = __expf(Sr[j] * SCALE - sm); sum += e; }
            Sr[j] = e;
        }
#pragma unroll
        for (int off = 16; off; off >>= 1) sum += __shfl_xor(sum, off);
        float inv = 1.f / sum;
        for (int j = lane; j <= jmax; j += 32) Sr[j] *= inv;
    }
    __syncthreads();

    // Phase 4: O(16x64) = P(16x2048) @ V(2048x64); waves split (d-tile, K-half)
    int ntile = wave & 3, kh = wave >> 2;
    const bf16* vbase = vvT + ((b * NH + hn) * DH + ntile * 16) * KLEN;
    v8f acc = {};
    for (int ks = 0; ks < 32; ++ks) {
        int k0 = kh * 1024 + ks * 32;
        v16bf a = ld_a16_f32(S + k0, KLEN);
        v16bf bb = ld_bT16(vbase + k0, KLEN);
        acc = wmma_bf16(a, bb, acc);
    }
    {
        int n16 = lane & 15, h = lane >> 4;
#pragma unroll
        for (int p = 0; p < 8; ++p)
            atomicAdd(&O[(p + 8 * h) * DH + ntile * 16 + n16], acc[p]);
    }
    __syncthreads();

    // Phase 5: write attn_vec rows in (i, b, n*64+d) layout (bf16 for final GEMM)
    for (int e = tid; e < 16 * DH; e += 256) {
        int li = e >> 6, d = e & 63;
        av[((i0 + li) * BSZ + b) * (NH * DH) + hn * DH + d] = (bf16)O[e];
    }
}

// ---------------- output projection: av[4096,1024] @ Wo^T -> fp32 -------------
__global__ __launch_bounds__(256) void gemm_out(const bf16* __restrict__ av,
                                                const bf16* __restrict__ Wo,
                                                float* __restrict__ aout) {
    const int NT = DMODEL / 16;
    int wave = threadIdx.x >> 5, lane = threadIdx.x & 31;
    int tile = blockIdx.x * 8 + wave;
    int mt = tile / NT, nt = tile % NT;
    v8f acc = {};
    for (int k0 = 0; k0 < DMODEL; k0 += 32) {
        v16bf a = ld_a16(av + mt * 16 * DMODEL + k0, DMODEL);
        v16bf b = ld_bT16(Wo + nt * 16 * DMODEL + k0, DMODEL);
        acc = wmma_bf16(a, b, acc);
    }
    int n16 = lane & 15, h = lane >> 4;
#pragma unroll
    for (int p = 0; p < 8; ++p)
        aout[(mt * 16 + p + 8 * h) * DMODEL + nt * 16 + n16] = acc[p];
}

// ---------------- residual + LayerNorm ----------------------------------------
__global__ __launch_bounds__(256) void ln_kernel(
    const float* __restrict__ w, const float* __restrict__ aout,
    const float* __restrict__ gamma, const float* __restrict__ beta,
    float* __restrict__ out) {
    __shared__ float red[8];
    int row = blockIdx.x, tid = threadIdx.x, lane = tid & 31, wave = tid >> 5;
    const float* wp = w + (size_t)row * DMODEL;
    const float* ap = aout + (size_t)row * DMODEL;
    float x[4];
    float s = 0.f;
#pragma unroll
    for (int t = 0; t < 4; ++t) {
        int c = tid + t * 256;
        x[t] = wp[c] + ap[c];
        s += x[t];
    }
#pragma unroll
    for (int off = 16; off; off >>= 1) s += __shfl_xor(s, off);
    if (lane == 0) red[wave] = s;
    __syncthreads();
    s = 0.f;
#pragma unroll
    for (int t = 0; t < 8; ++t) s += red[t];
    float mu = s * (1.0f / DMODEL);
    __syncthreads();
    float v = 0.f;
#pragma unroll
    for (int t = 0; t < 4; ++t) {
        float d = x[t] - mu;
        v += d * d;
    }
#pragma unroll
    for (int off = 16; off; off >>= 1) v += __shfl_xor(v, off);
    if (lane == 0) red[wave] = v;
    __syncthreads();
    v = 0.f;
#pragma unroll
    for (int t = 0; t < 8; ++t) v += red[t];
    float rstd = rsqrtf(v * (1.0f / DMODEL) + LN_EPS);
#pragma unroll
    for (int t = 0; t < 4; ++t) {
        int c = tid + t * 256;
        out[(size_t)row * DMODEL + c] = (x[t] - mu) * rstd * gamma[c] + beta[c];
    }
}

// ---------------- host launcher ------------------------------------------------
extern "C" void kernel_launch(void* const* d_in, const int* in_sizes, int n_in,
                              void* d_out, int out_size, void* d_ws, size_t ws_size,
                              hipStream_t stream) {
    const float* w    = (const float*)d_in[0];
    const float* r    = (const float*)d_in[1];
    const float* rwb  = (const float*)d_in[2];
    const float* rrb  = (const float*)d_in[3];
    const float* mems = (const float*)d_in[4];
    // d_in[5] attn_mask: computed analytically (j > i + MLEN), not read
    const float* Wqkv = (const float*)d_in[6];
    const float* Wr   = (const float*)d_in[7];
    const float* Wo   = (const float*)d_in[8];
    const float* gam  = (const float*)d_in[9];
    const float* bet  = (const float*)d_in[10];
    float* out = (float*)d_out;

    char* p = (char*)d_ws;
    auto carve = [&](size_t bytes) {
        char* q = p;
        p += (bytes + 255) & ~(size_t)255;
        return q;
    };
    bf16* catb  = (bf16*)carve((size_t)KLEN * BSZ * DMODEL * 2);       // 16 MiB
    bf16* rbf   = (bf16*)carve((size_t)KLEN * DMODEL * 2);             //  4 MiB
    bf16* Wqkvb = (bf16*)carve((size_t)3 * DMODEL * DMODEL * 2);       //  6 MiB
    bf16* Wrb   = (bf16*)carve((size_t)DMODEL * DMODEL * 2);           //  2 MiB
    bf16* Wob   = (bf16*)carve((size_t)DMODEL * DMODEL * 2);           //  2 MiB
    bf16* qwb   = (bf16*)carve((size_t)BSZ * NH * QLEN * DH * 2);      //  8 MiB
    bf16* qrb   = (bf16*)carve((size_t)BSZ * NH * QLEN * DH * 2);      //  8 MiB
    bf16* kkb   = (bf16*)carve((size_t)BSZ * NH * KLEN * DH * 2);      // 16 MiB
    bf16* vvTb  = (bf16*)carve((size_t)BSZ * NH * DH * KLEN * 2);      // 16 MiB
    bf16* rkb   = (bf16*)carve((size_t)NH * KLEN * DH * 2);            //  4 MiB
    bf16* avb   = (bf16*)carve((size_t)QLEN * BSZ * NH * DH * 2);      //  8 MiB
    float* aoutw = (float*)carve((size_t)QLEN * BSZ * DMODEL * 4);     // 16 MiB

    // 1) bf16 staging
    cvt_f32_bf16<<<4096, 256, 0, stream>>>(mems, catb, MLEN * BSZ * DMODEL);
    cvt_f32_bf16<<<4096, 256, 0, stream>>>(w, catb + MLEN * BSZ * DMODEL,
                                           QLEN * BSZ * DMODEL);
    cvt_f32_bf16<<<2048, 256, 0, stream>>>(r, rbf, KLEN * DMODEL);
    cvt_f32_bf16<<<4096, 256, 0, stream>>>(Wqkv, Wqkvb, 3 * DMODEL * DMODEL);
    cvt_f32_bf16<<<2048, 256, 0, stream>>>(Wr, Wrb, DMODEL * DMODEL);
    cvt_f32_bf16<<<2048, 256, 0, stream>>>(Wo, Wob, DMODEL * DMODEL);

    // 2) projections (WMMA)
    gemm_qkv<<<12288, 256, 0, stream>>>(catb, Wqkvb, rwb, rrb, qwb, qrb, kkb, vvTb);
    gemm_r<<<1024, 256, 0, stream>>>(rbf, Wrb, rkb);

    // 3) fused attention (WMMA + LDS-resident scores)
    attn_kernel<<<BSZ * NH * (QLEN / 16), 256, 0, stream>>>(qwb, qrb, kkb, rkb,
                                                            vvTb, avb);

    // 4) output projection + residual LayerNorm
    gemm_out<<<2048, 256, 0, stream>>>(avb, Wob, aoutw);
    ln_kernel<<<QLEN * BSZ, 256, 0, stream>>>(w, aoutw, gam, bet, out);
}
